// Pyg_Net_31104153158262
// MI455X (gfx1250) — matrix operand order repaired and checked by reading, not compiled, not run on hardware
//
#include <hip/hip_runtime.h>
#include <hip/hip_bf16.h>

typedef __attribute__((ext_vector_type(16))) _Float16 v16h;
typedef __attribute__((ext_vector_type(8)))  _Float16 v8h;
typedef __attribute__((ext_vector_type(8)))  float    v8f;

#define N_NODES 100000
#define N_EDGES 1600000

constexpr int A_NONE = 0, A_CLIP = 1, A_RELU = 2, A_BN = 3;
constexpr int E_SELF = 0, E_BIAS_RELU = 1, E_BIAS = 2;

__device__ __forceinline__ void atomAddF(float* p, float v) {
    __hip_atomic_fetch_add(p, v, __ATOMIC_RELAXED, __HIP_MEMORY_SCOPE_AGENT);
}

// ---------------- degree / dinv ----------------
__global__ void k_degree(const int* __restrict__ dst, float* __restrict__ deg, int E) {
    int i = blockIdx.x * blockDim.x + threadIdx.x;
    if (i < E) atomAddF(&deg[dst[i]], 1.0f);
}

__global__ void k_dinv(const float* __restrict__ deg, float* __restrict__ dinv, int n) {
    int i = blockIdx.x * blockDim.x + threadIdx.x;
    if (i < n) dinv[i] = rsqrtf(deg[i] + 1.0f);
}

// ------------- pack weight: f32 [K x M] -> f16 transposed padded [MPad x KPad] -------------
__global__ void k_pack_w(const float* __restrict__ W, int K, int M,
                         _Float16* __restrict__ Wt, int KPad, int MPad) {
    int idx = blockIdx.x * blockDim.x + threadIdx.x;
    if (idx >= KPad * MPad) return;
    int m = idx / KPad, k = idx - m * KPad;
    float v = (m < M && k < K) ? W[(size_t)k * M + m] : 0.0f;
    Wt[idx] = (_Float16)v;
}

// ---------------- WMMA fragment types ----------------
union AFrag { v16h v; v8h h[2]; };

__device__ __forceinline__ v8f wmma_f16(const v16h& a, const v16h& b, const v8f& c) {
    return __builtin_amdgcn_wmma_f32_16x16x32_f16(false, a, false, b, (short)0, c, false, false);
}

// ---------------- WMMA GEMM, fully specialized ----------------
// Block: 256 threads = 8 waves. Covers 32 rows (2 M-tiles) x 256 cols (8 waves x 2 N-tiles).
// grid.x = N/32 = 3125 (exact). grid.y = 1 (MoutPad <= 256 for all layers).
// Per k-step each wave: 2 A-frags (LDS), 2 B-frags (global, L2-resident), 4 WMMAs.
template<int KACT, int KPAD, int AMODE, int EMODE>
__global__ __launch_bounds__(256) void k_gemm(
    const float* __restrict__ A, int lda,
    const _Float16* __restrict__ Wt, int MoutPad, int Mact,
    const float* __restrict__ bias,
    const float* __restrict__ bn_a, const float* __restrict__ bn_b,
    const float* __restrict__ dinv,
    float* __restrict__ Out, float* __restrict__ Agg)
{
    __shared__ _Float16 Alds[32 * 40];   // 32 rows x 32 k, row stride 40 (conflict-free frags)

    const int tid  = threadIdx.x;
    const int lane = tid & 31;
    const int wave = tid >> 5;
    const int half = lane >> 4;
    const int lcol = lane & 15;
    const int row0 = blockIdx.x * 32;
    const int n0   = wave * 16;           // col-tile 0
    const int n1   = (wave + 8) * 16;     // col-tile 1
    const bool act0 = (n0 < MoutPad);
    const bool act1 = (n1 < MoutPad);

    v8f acc00 = {}, acc10 = {};   // [rowtile 0/1][coltile 0]
    v8f acc01 = {}, acc11 = {};   // [rowtile 0/1][coltile 1]

    #pragma unroll
    for (int kk = 0; kk < KPAD; kk += 32) {
        __syncthreads();
        // stage 32x32 A tile: 1024 elements, 4 per thread, fused transform -> f16
        #pragma unroll
        for (int t = 0; t < 4; ++t) {
            const int idx = tid + t * 256;
            const int r = idx >> 5;
            const int c = idx & 31;
            const int k = kk + c;          // kk is compile-time after unroll
            float v = 0.0f;
            if (k < KACT) {
                v = A[(size_t)(row0 + r) * lda + k];
                if constexpr (AMODE == A_CLIP) v = fminf(fmaxf(v, -0.4f), 0.4f);
                if constexpr (AMODE == A_RELU) v = fmaxf(v, 0.0f);
                if constexpr (AMODE == A_BN)   v = fmaxf(v * bn_a[k] + bn_b[k], 0.0f);
            }
            Alds[r * 40 + c] = (_Float16)v;
        }
        __syncthreads();

        // A fragments (contiguous layout: lanes 0-15 K=0..15, lanes 16-31 K=16..31)
        AFrag a0, a1;
        const _Float16* ap0 = &Alds[lcol * 40 + half * 16];
        a0.h[0] = *(const v8h*)(ap0);
        a0.h[1] = *(const v8h*)(ap0 + 8);
        const _Float16* ap1 = ap0 + 16 * 40;
        a1.h[0] = *(const v8h*)(ap1);
        a1.h[1] = *(const v8h*)(ap1 + 8);

        if (act0) {
            const v16h b0 = *(const v16h*)(Wt + (size_t)(n0 + lcol) * KPAD + kk + half * 16);
            acc00 = wmma_f16(a0.v, b0, acc00);
            acc10 = wmma_f16(a1.v, b0, acc10);
        }
        if (act1) {
            const v16h b1 = *(const v16h*)(Wt + (size_t)(n1 + lcol) * KPAD + kk + half * 16);
            acc01 = wmma_f16(a0.v, b1, acc01);
            acc11 = wmma_f16(a1.v, b1, acc11);
        }
    }

    // ---- epilogue ----
    #pragma unroll
    for (int j = 0; j < 2; ++j) {
        const bool act = j ? act1 : act0;
        if (!act) continue;
        const int col = (j ? n1 : n0) + lcol;
        if (col >= Mact) continue;
        const float b = bias ? bias[col] : 0.0f;
        #pragma unroll
        for (int rt = 0; rt < 2; ++rt) {
            const v8f acc = j ? (rt ? acc11 : acc01) : (rt ? acc10 : acc00);
            #pragma unroll
            for (int r = 0; r < 8; ++r) {
                const int row = row0 + rt * 16 + r + 8 * half;
                const float c = acc[r];
                const size_t o = (size_t)row * Mact + col;
                if constexpr (EMODE == E_SELF) {       // GCN: keep xw, agg = xw*dinv^2 + bias
                    Out[o] = c;
                    const float di = dinv[row];
                    Agg[o] = c * di * di + b;
                } else if constexpr (EMODE == E_BIAS_RELU) {
                    Out[o] = fmaxf(c + b, 0.0f);
                } else {                               // E_BIAS
                    Out[o] = c + b;
                }
            }
        }
    }
}

// ---------------- edge scatter: one wave per edge ----------------
__global__ __launch_bounds__(256) void k_scatter(
    const int* __restrict__ src, const int* __restrict__ dst,
    const float* __restrict__ dinv,
    const float* __restrict__ xw, float* __restrict__ agg,
    int E, int M)
{
    const int e    = (blockIdx.x * blockDim.x + threadIdx.x) >> 5;
    const int lane = threadIdx.x & 31;
    if (e >= E) return;
    const int s = src[e], d = dst[e];
    const float norm = dinv[s] * dinv[d];
    const float* xs = xw + (size_t)s * M;
    float* ad = agg + (size_t)d * M;
    for (int m = lane; m < M; m += 32)
        atomAddF(&ad[m], norm * xs[m]);
}

// ---------------- batchnorm stats ----------------
__global__ __launch_bounds__(128) void k_bn_reduce(
    const float* __restrict__ h, float* __restrict__ sum, float* __restrict__ sumsq, int n)
{
    const int col = threadIdx.x;                    // 128 cols
    const int r0 = blockIdx.x * 256;
    const int r1 = min(r0 + 256, n);
    float s = 0.0f, ss = 0.0f;
    for (int r = r0; r < r1; ++r) {
        float v = h[(size_t)r * 128 + col];
        s += v; ss += v * v;
    }
    atomAddF(&sum[col], s);
    atomAddF(&sumsq[col], ss);
}

__global__ void k_bn_final(const float* __restrict__ sum, const float* __restrict__ sumsq,
                           const float* __restrict__ gamma, const float* __restrict__ beta,
                           float* __restrict__ bn_a, float* __restrict__ bn_b, int n)
{
    const int c = threadIdx.x;
    const float mu  = sum[c] / (float)n;
    const float var = sumsq[c] / (float)n - mu * mu;
    const float inv = rsqrtf(var + 1e-5f);
    const float a = gamma[c] * inv;
    bn_a[c] = a;
    bn_b[c] = beta[c] - mu * a;
}

// ---------------- log_softmax over 19 labels ----------------
__global__ void k_logsoftmax(const float* __restrict__ logits, float* __restrict__ out, int n) {
    const int r = blockIdx.x * blockDim.x + threadIdx.x;
    if (r >= n) return;
    const float* L = logits + (size_t)r * 19;
    float m = -1e30f;
    #pragma unroll
    for (int j = 0; j < 19; ++j) m = fmaxf(m, L[j]);
    float s = 0.0f;
    #pragma unroll
    for (int j = 0; j < 19; ++j) s += __expf(L[j] - m);
    const float ls = __logf(s) + m;
    #pragma unroll
    for (int j = 0; j < 19; ++j) out[(size_t)r * 19 + j] = L[j] - ls;
}

extern "C" void kernel_launch(void* const* d_in, const int* in_sizes, int n_in,
                              void* d_out, int out_size, void* d_ws, size_t ws_size,
                              hipStream_t stream) {
    (void)in_sizes; (void)n_in; (void)out_size; (void)ws_size;
    const float* x     = (const float*)d_in[0];
    const int*   ei    = (const int*)d_in[1];
    const float* W1    = (const float*)d_in[2];  const float* b1    = (const float*)d_in[3];
    const float* W2    = (const float*)d_in[4];  const float* b2    = (const float*)d_in[5];
    const float* W3    = (const float*)d_in[6];  const float* b3    = (const float*)d_in[7];
    const float* fc1W  = (const float*)d_in[8];  const float* fc1b  = (const float*)d_in[9];
    const float* fc2aW = (const float*)d_in[10]; const float* fc2ab = (const float*)d_in[11];
    const float* gamma = (const float*)d_in[12]; const float* beta  = (const float*)d_in[13];
    const float* fc2bW = (const float*)d_in[14]; const float* fc2bb = (const float*)d_in[15];
    const int* srcI = ei;
    const int* dstI = ei + N_EDGES;

    // ---- workspace suballocation (256B aligned) ----
    char* ws = (char*)d_ws;
    size_t off = 0;
    auto alloc = [&](size_t bytes) -> void* {
        void* p = ws + off;
        off = (off + bytes + 255) & ~(size_t)255;
        return p;
    };
    float*    deg   = (float*)alloc((size_t)N_NODES * 4);
    float*    dinv  = (float*)alloc((size_t)N_NODES * 4);
    float*    bsum  = (float*)alloc(128 * 4);
    float*    bsq   = (float*)alloc(128 * 4);
    float*    bn_a  = (float*)alloc(128 * 4);
    float*    bn_b  = (float*)alloc(128 * 4);
    _Float16* W1t   = (_Float16*)alloc((size_t)112 * 128 * 2);
    _Float16* W2t   = (_Float16*)alloc((size_t)112 * 128 * 2);
    _Float16* W3t   = (_Float16*)alloc((size_t)256 * 128 * 2);
    _Float16* fc1t  = (_Float16*)alloc((size_t)256 * 256 * 2);
    _Float16* fc2at = (_Float16*)alloc((size_t)128 * 256 * 2);
    _Float16* fc2bt = (_Float16*)alloc((size_t)32  * 128 * 2);
    float*    XW    = (float*)alloc((size_t)N_NODES * 256 * 4);
    float*    P0    = (float*)alloc((size_t)N_NODES * 256 * 4);
    float*    P1    = (float*)alloc((size_t)N_NODES * 256 * 4);

    const int GB = 3125;  // 100000 / 32 row-blocks

    // ---- degree / dinv ----
    hipMemsetAsync(deg, 0, (size_t)N_NODES * 4, stream);
    k_degree<<<(N_EDGES + 255) / 256, 256, 0, stream>>>(dstI, deg, N_EDGES);
    k_dinv<<<(N_NODES + 255) / 256, 256, 0, stream>>>(deg, dinv, N_NODES);

    // ---- pack weights to f16 transposed/padded ----
    k_pack_w<<<(112 * 128 + 255) / 256, 256, 0, stream>>>(W1,   100, 100, W1t,   128, 112);
    k_pack_w<<<(112 * 128 + 255) / 256, 256, 0, stream>>>(W2,   100, 100, W2t,   128, 112);
    k_pack_w<<<(256 * 128 + 255) / 256, 256, 0, stream>>>(W3,   100, 256, W3t,   128, 256);
    k_pack_w<<<(256 * 256 + 255) / 256, 256, 0, stream>>>(fc1W, 256, 256, fc1t,  256, 256);
    k_pack_w<<<(128 * 256 + 255) / 256, 256, 0, stream>>>(fc2aW,256, 128, fc2at, 256, 128);
    k_pack_w<<<(32  * 128 + 255) / 256, 256, 0, stream>>>(fc2bW,128, 19,  fc2bt, 128, 32);

    const int sgrid = (N_EDGES + 7) / 8;   // 8 waves (edges) per 256-thread block

    // ---- GCN layer 1: clip(x) @ W1 ; agg=P0 ----
    k_gemm<100, 128, A_CLIP, E_SELF><<<GB, 256, 0, stream>>>(
        x, 100, W1t, 112, 100, b1, nullptr, nullptr, dinv, XW, P0);
    k_scatter<<<sgrid, 256, 0, stream>>>(srcI, dstI, dinv, XW, P0, N_EDGES, 100);

    // ---- GCN layer 2: relu(P0) @ W2 ; agg=P1 ----
    k_gemm<100, 128, A_RELU, E_SELF><<<GB, 256, 0, stream>>>(
        P0, 100, W2t, 112, 100, b2, nullptr, nullptr, dinv, XW, P1);
    k_scatter<<<sgrid, 256, 0, stream>>>(srcI, dstI, dinv, XW, P1, N_EDGES, 100);

    // ---- GCN layer 3: relu(P1) @ W3 (100->256) ; agg=P0 ----
    k_gemm<100, 128, A_RELU, E_SELF><<<GB, 256, 0, stream>>>(
        P1, 100, W3t, 256, 256, b3, nullptr, nullptr, dinv, XW, P0);
    k_scatter<<<sgrid, 256, 0, stream>>>(srcI, dstI, dinv, XW, P0, N_EDGES, 256);

    // ---- fc1: relu(P0) @ fc1W + b, relu -> P1 ----
    k_gemm<256, 256, A_RELU, E_BIAS_RELU><<<GB, 256, 0, stream>>>(
        P0, 256, fc1t, 256, 256, fc1b, nullptr, nullptr, dinv, P1, nullptr);

    // ---- fc2a: P1 @ fc2aW + b -> XW (N x 128, pre-BN) ----
    k_gemm<256, 256, A_NONE, E_BIAS><<<GB, 256, 0, stream>>>(
        P1, 256, fc2at, 128, 128, fc2ab, nullptr, nullptr, dinv, XW, nullptr);

    // ---- batchnorm stats (train-mode batch stats) ----
    hipMemsetAsync(bsum, 0, 128 * 4, stream);
    hipMemsetAsync(bsq,  0, 128 * 4, stream);
    k_bn_reduce<<<(N_NODES + 255) / 256, 128, 0, stream>>>(XW, bsum, bsq, N_NODES);
    k_bn_final<<<1, 128, 0, stream>>>(bsum, bsq, gamma, beta, bn_a, bn_b, N_NODES);

    // ---- fc2b: relu(BN(XW)) @ fc2bW + b -> P0 (N x 19) ; BN+relu fused into A load ----
    k_gemm<128, 128, A_BN, E_BIAS><<<GB, 256, 0, stream>>>(
        XW, 128, fc2bt, 32, 19, fc2bb, bn_a, bn_b, dinv, P0, nullptr);

    // ---- log_softmax -> d_out ----
    k_logsoftmax<<<(N_NODES + 255) / 256, 256, 0, stream>>>(P0, (float*)d_out, N_NODES);
}
